// Model_interra_39522289058026
// MI455X (gfx1250) — matrix operand order repaired and checked by reading, not compiled, not run on hardware
//
#include <hip/hip_runtime.h>
#include <hip/hip_bf16.h>
#include <stdint.h>

// ---------------------------------------------------------------------------
// Problem constants (ViT-Base token-swap alignment, see reference)
// ---------------------------------------------------------------------------
#define DIMV   768
#define BATCH  128
#define PTOK   256      // patch tokens per sample (row 0 is CLS, untouched)
#define ROWS   257      // P + 1
#define LAYERS 12
#define SWAPS  20
#define EPSLN  1e-5f

typedef __attribute__((ext_vector_type(16))) _Float16 v16h;
typedef __attribute__((ext_vector_type(8)))  float    v8f;

union FragU {
    uint32_t u[8];
    v16h     v;
};

__device__ __forceinline__ uint32_t pack_f16x2(float a, float b) {
    union { _Float16 h[2]; uint32_t u; } p;
    p.h[0] = (_Float16)a;   // RNE convert
    p.h[1] = (_Float16)b;
    return p.u;
}

// ---------------------------------------------------------------------------
// Kernel 1: per-row LayerNorm -> f16 workspace; also ||rgb_j||^2 (fp32).
// One wave32 per token row; 8 rows per 256-thread block.
// rows [0, 32768)  -> rgb stream (x1 patches)
// rows [32768, 65536) -> sket stream (x2 patches)
// ---------------------------------------------------------------------------
__global__ __launch_bounds__(256) void ln_kernel(
    const float* __restrict__ y1, const float* __restrict__ y2,
    const float* __restrict__ rgb_w, const float* __restrict__ rgb_b,
    const float* __restrict__ skt_w, const float* __restrict__ skt_b,
    uint16_t* __restrict__ rgbLN, uint16_t* __restrict__ sktLN,
    float* __restrict__ s_rgb)
{
    const int wave = threadIdx.x >> 5;
    const int lane = threadIdx.x & 31;
    const int rid  = blockIdx.x * 8 + wave;          // 0 .. 65535
    const bool isRgb = rid < (BATCH * PTOK);
    const int  row   = isRgb ? rid : rid - BATCH * PTOK;  // 0 .. 32767
    const int  b = row >> 8;
    const int  p = row & 255;

    const float* src = (isRgb ? y1 : y2) + (size_t)(b * ROWS + 1 + p) * DIMV;
    const float* wv  = isRgb ? rgb_w : skt_w;
    const float* bv  = isRgb ? rgb_b : skt_b;
    uint16_t*    dst = (isRgb ? rgbLN : sktLN) + (size_t)row * DIMV;

    // 768 floats = 192 float4; 32 lanes x 6 float4 each, coalesced.
    float4 xv[6];
    float s = 0.f, sq = 0.f;
    const float4* s4 = reinterpret_cast<const float4*>(src);
#pragma unroll
    for (int c = 0; c < 6; ++c) {
        float4 v = s4[c * 32 + lane];
        xv[c] = v;
        s  += v.x + v.y + v.z + v.w;
        sq += v.x * v.x + v.y * v.y + v.z * v.z + v.w * v.w;
    }
#pragma unroll
    for (int o = 16; o >= 1; o >>= 1) {
        s  += __shfl_xor(s,  o, 32);
        sq += __shfl_xor(sq, o, 32);
    }
    const float mean = s * (1.f / DIMV);
    const float var  = sq * (1.f / DIMV) - mean * mean;
    const float rs   = rsqrtf(var + EPSLN);

    const float4* w4 = reinterpret_cast<const float4*>(wv);
    const float4* b4 = reinterpret_cast<const float4*>(bv);
    uint2* d2 = reinterpret_cast<uint2*>(dst);
    float ss = 0.f;
#pragma unroll
    for (int c = 0; c < 6; ++c) {
        float4 ww = w4[c * 32 + lane];
        float4 bb = b4[c * 32 + lane];
        float y0 = (xv[c].x - mean) * rs * ww.x + bb.x;
        float y1v = (xv[c].y - mean) * rs * ww.y + bb.y;
        float y2v = (xv[c].z - mean) * rs * ww.z + bb.z;
        float y3 = (xv[c].w - mean) * rs * ww.w + bb.w;
        ss += y0 * y0 + y1v * y1v + y2v * y2v + y3 * y3;
        uint2 o;
        o.x = pack_f16x2(y0, y1v);
        o.y = pack_f16x2(y2v, y3);
        d2[c * 32 + lane] = o;
    }
    if (isRgb) {
#pragma unroll
        for (int o = 16; o >= 1; o >>= 1) ss += __shfl_xor(ss, o, 32);
        if (lane == 0) s_rgb[row] = ss;
    }
}

// ---------------------------------------------------------------------------
// Kernel 2: per (sample, i-tile of 16 sketch rows): G = skt_tile @ rgb^T over
// all 256 rgb rows, K=768, via v_wmma_f32_16x16x32_f16; fused argmin_j of
// (||rgb_j||^2 - 2*G[i,j])  (== argmin of clipped/sqrt distance).
// 8 waves per block; wave w owns j-tiles {2w, 2w+1}.
//
// Fragment layouts (ISA 7.12.2, 16-bit, wave32):
//   A (16x32, skt rows):  lane L -> row (L&15); VGPR v dword = K pair at
//       K = 2*(v&3) + 16*(v>>2) + 8*(L>>4)  => two b128 loads @ bytes
//       kk*64 + 16*half  and  kk*64 + 32 + 16*half  within the row.
//   B (32x16 = rgb^T):    lane L -> col n = rgb row (L&15); same K order
//       as A => two b128
//       loads @ bytes kk*64 + 16*half  and  +32.
//   C/D (16x16 f32): VGPR r -> i-row r + 8*(L>>4), col j = L&15.
// ---------------------------------------------------------------------------
__global__ __launch_bounds__(256) void gemm_argmin_kernel(
    const uint16_t* __restrict__ rgbLN, const uint16_t* __restrict__ sktLN,
    const float* __restrict__ s_rgb, int* __restrict__ idx)
{
    __shared__ unsigned long long best[16];

    const int b    = blockIdx.x >> 4;
    const int it   = blockIdx.x & 15;
    const int wave = threadIdx.x >> 5;
    const int lane = threadIdx.x & 31;
    const int half = lane >> 4;
    const int l16  = lane & 15;

    if (threadIdx.x < 16) best[threadIdx.x] = ~0ULL;
    __syncthreads();

    const char* Abase = (const char*)(sktLN + ((size_t)b * PTOK + it * 16 + l16) * DIMV);

    unsigned long long wbest[8];
#pragma unroll
    for (int r = 0; r < 8; ++r) wbest[r] = ~0ULL;

    for (int jt2 = 0; jt2 < 2; ++jt2) {
        const int jt = wave * 2 + jt2;
        const char* Bbase = (const char*)(rgbLN + ((size_t)b * PTOK + jt * 16 + l16) * DIMV);

        v8f acc = {};
        for (int kk = 0; kk < 24; ++kk) {
            FragU a, bf;
            const char* ap = Abase + kk * 64 + 16 * half;
            *(uint4*)&a.u[0]  = *(const uint4*)(ap);
            *(uint4*)&a.u[4]  = *(const uint4*)(ap + 32);
            const char* bp = Bbase + kk * 64 + 16 * half;
            *(uint4*)&bf.u[0] = *(const uint4*)(bp);
            *(uint4*)&bf.u[4] = *(const uint4*)(bp + 32);
            acc = __builtin_amdgcn_wmma_f32_16x16x32_f16(
                false, a.v, false, bf.v, (short)0, acc, false, false);
        }

        const int   j   = jt * 16 + l16;
        const float srj = s_rgb[b * PTOK + j];
#pragma unroll
        for (int r = 0; r < 8; ++r) {
            float score = srj - 2.0f * acc[r];          // monotone in distance
            uint32_t k = __float_as_uint(score);
            k = (k & 0x80000000u) ? ~k : (k | 0x80000000u);  // sortable key
            unsigned long long cand =
                ((unsigned long long)k << 32) | (uint32_t)j; // tie -> min j
#pragma unroll
            for (int o = 8; o >= 1; o >>= 1) {               // within 16-lane half
                unsigned long long other = __shfl_xor(cand, o, 32);
                cand = other < cand ? other : cand;
            }
            wbest[r] = cand < wbest[r] ? cand : wbest[r];
        }
    }

#pragma unroll
    for (int r = 0; r < 8; ++r) {
        if (l16 == 0) atomicMin(&best[r + 8 * half], wbest[r]); // ds_min_u64
    }
    __syncthreads();
    if (threadIdx.x < 16)
        idx[b * PTOK + it * 16 + threadIdx.x] =
            (int)(best[threadIdx.x] & 0xFFFFFFFFu);
}

// ---------------------------------------------------------------------------
// Kernel 3: 20 sequential cross-stream row swaps per sample.
// Element e of any row is always handled by thread e%256 -> per-thread
// program order carries the sequential dependency; no barriers needed.
// ---------------------------------------------------------------------------
__global__ __launch_bounds__(256) void swap_kernel(
    float* __restrict__ y1, float* __restrict__ y2,
    const int* __restrict__ seeds, const int* __restrict__ idx, int layer)
{
    const int b = blockIdx.x;
    const int t = threadIdx.x;
    float* p1 = y1 + (size_t)b * ROWS * DIMV + DIMV;   // patch tokens of x1
    float* p2 = y2 + (size_t)b * ROWS * DIMV + DIMV;   // patch tokens of x2

    for (int s = 0; s < SWAPS; ++s) {
        const int seed = seeds[layer * SWAPS + s];
        const int sim  = idx[b * PTOK + seed];
        float* r1 = p1 + (size_t)sim  * DIMV;
        float* r2 = p2 + (size_t)seed * DIMV;
#pragma unroll
        for (int c = 0; c < 3; ++c) {
            const int e = t + c * 256;
            float h1 = r1[e];
            float h2 = r2[e];
            r2[e] = h1;
            r1[e] = h2;
        }
    }
}

// ---------------------------------------------------------------------------
// Host launcher
// ---------------------------------------------------------------------------
extern "C" void kernel_launch(void* const* d_in, const int* in_sizes, int n_in,
                              void* d_out, int out_size, void* d_ws, size_t ws_size,
                              hipStream_t stream)
{
    const float* x1    = (const float*)d_in[0];
    const float* x2    = (const float*)d_in[1];
    const float* rgb_w = (const float*)d_in[2];
    const float* rgb_b = (const float*)d_in[3];
    const float* skt_w = (const float*)d_in[4];
    const float* skt_b = (const float*)d_in[5];
    const int*   seeds = (const int*)d_in[6];

    const size_t N = (size_t)BATCH * ROWS * DIMV;     // elems per stream
    float* y1 = (float*)d_out;
    float* y2 = y1 + N;

    // Outputs start as copies of the inputs; all layers operate in-place.
    hipMemcpyAsync(y1, x1, N * sizeof(float), hipMemcpyDeviceToDevice, stream);
    hipMemcpyAsync(y2, x2, N * sizeof(float), hipMemcpyDeviceToDevice, stream);

    // Workspace carve-up (~96.3 MiB), all 256B-aligned regions.
    const size_t lnBytes = (size_t)BATCH * PTOK * DIMV * sizeof(uint16_t); // 48 MiB
    char* ws = (char*)d_ws;
    uint16_t* rgbLN = (uint16_t*)ws;
    uint16_t* sktLN = (uint16_t*)(ws + lnBytes);
    float*    s_rgb = (float*)(ws + 2 * lnBytes);
    int*      idx   = (int*)(ws + 2 * lnBytes + (size_t)BATCH * PTOK * sizeof(float));

    for (int layer = 0; layer < LAYERS; ++layer) {
        ln_kernel<<<(2 * BATCH * PTOK) / 8, 256, 0, stream>>>(
            y1, y2, rgb_w, rgb_b, skt_w, skt_b, rgbLN, sktLN, s_rgb);
        gemm_argmin_kernel<<<BATCH * 16, 256, 0, stream>>>(
            rgbLN, sktLN, s_rgb, idx);
        swap_kernel<<<BATCH, 256, 0, stream>>>(y1, y2, seeds, idx, layer);
    }
}
